// HandWritingSynthRNN_44169443672918
// MI455X (gfx1250) — matrix operand order repaired and checked by reading, not compiled, not run on hardware
//
#include <hip/hip_runtime.h>
#include <hip/hip_bf16.h>

// ---------------- problem constants ----------------
#define T_STEPS 800
#define BATCH   128
#define HID     400
#define KMIX    10
#define NGAUSS  20
#define NCHARS  57
#define ULEN    80
#define KFC     0.05f

#define G4H     1600          // 4*H gate columns
#define K1PAD   480           // [x(3) win(57) h1(400) pad(20)]        -> 15 k-tiles
#define K2PAD   864           // [h1(400) x(3) win(57) h2(400) pad(4)] -> 27 k-tiles
#define KOUT    800           // [h1 h2]                               -> 25 k-tiles
#define YCOLS   121

// ---------------- launch geometry ----------------
#define NWG      32           // persistent workgroups (all co-resident)
#define GATE_WGS 28           // WGs doing gates1 in slot A; the rest run the output head

typedef __bf16 bf16_t;
typedef bf16_t v16bf __attribute__((ext_vector_type(16)));
typedef bf16_t v8bf  __attribute__((ext_vector_type(8)));
typedef float  v8f   __attribute__((ext_vector_type(8)));

// Global-memory (address space 1) element types: forces global_load/global_store
// instead of flat_* (flat ties up DScnt in addition to LOADcnt, ISA 10.1.1).
typedef __attribute__((address_space(1))) float        gf32;
typedef __attribute__((address_space(1))) const float  gcf32;
typedef __attribute__((address_space(1))) bf16_t       gbf;
typedef __attribute__((address_space(1))) const bf16_t gcbf;
typedef __attribute__((address_space(1))) const v8bf   gcv8bf;
typedef __attribute__((address_space(1))) const v16bf  gcv16bf;
typedef __attribute__((address_space(1))) const int    gci32;

// ---------------- small helpers ----------------
__device__ __host__ inline bf16_t f2bf(float f) {
  union { float f; unsigned u; } v; v.f = f;
  unsigned r = v.u + 0x7fffu + ((v.u >> 16) & 1u);     // round-to-nearest-even
  union { unsigned short s; bf16_t b; } o; o.s = (unsigned short)(r >> 16);
  return o.b;
}
__device__ inline float sigmf(float x) { return 1.f / (1.f + __expf(-x)); }

__device__ inline v8f wmma_bf16(v16bf a, v16bf b, v8f c) {
  return __builtin_amdgcn_wmma_f32_16x16x32_bf16(false, a, false, b, (short)0, c, false, false);
}

// A 16x32 bf16 fragment (wave32, contiguous k order used here, same as B): lanes 0-15 = rows, K 0..15;
// lanes 16-31 = K 16..31. `base` is already lane-adjusted to row*ldA + (lane&16?16:0).
__device__ inline v16bf load_a(gcbf* base, int kb) {
  union { v16bf v; v8bf h[2]; } u;
  u.h[0] = *(gcv8bf*)(base + kb * 32);
  u.h[1] = *(gcv8bf*)(base + kb * 32 + 8);
  return u.v;
}

// Two C tiles sharing one M tile (same A fragments, two independent WMMA chains).
__device__ inline void gemm_pair(const bf16_t* A, int ldA, const bf16_t* W, int ldW,
                                 float* G, int ldG, int mBase, int n0, int n1,
                                 int kTiles, int lane) {
  gcbf* ap = (gcbf*)A + (size_t)(mBase + (lane & 15)) * ldA + ((lane & 16) ? 16 : 0);
  gcbf* w0 = (gcbf*)W + (size_t)(n0 + (lane & 15)) * ldW + ((lane & 16) ? 16 : 0);
  gcbf* w1 = (gcbf*)W + (size_t)(n1 + (lane & 15)) * ldW + ((lane & 16) ? 16 : 0);
  v8f acc0 = {}, acc1 = {};
  for (int kb = 0; kb < kTiles; ++kb) {
    v16bf a  = load_a(ap, kb);
    v16bf b0 = *(gcv16bf*)(w0 + kb * 32);
    v16bf b1 = *(gcv16bf*)(w1 + kb * 32);
    acc0 = wmma_bf16(a, b0, acc0);
    acc1 = wmma_bf16(a, b1, acc1);
  }
  gf32* g = (gf32*)G;
  int rofs = (lane & 16) ? 8 : 0;
  int c0 = n0 + (lane & 15), c1 = n1 + (lane & 15);
#pragma unroll
  for (int rr = 0; rr < 8; ++rr) {
    g[(size_t)(mBase + rofs + rr) * ldG + c0] = acc0[rr];
    g[(size_t)(mBase + rofs + rr) * ldG + c1] = acc1[rr];
  }
}

__device__ inline void gemm_one(const bf16_t* A, int ldA, const bf16_t* W, int ldW,
                                float* G, int ldG, int mBase, int n0, int kTiles, int lane) {
  gcbf* ap = (gcbf*)A + (size_t)(mBase + (lane & 15)) * ldA + ((lane & 16) ? 16 : 0);
  gcbf* w0 = (gcbf*)W + (size_t)(n0 + (lane & 15)) * ldW + ((lane & 16) ? 16 : 0);
  v8f acc = {};
  for (int kb = 0; kb < kTiles; ++kb) {
    v16bf a  = load_a(ap, kb);
    v16bf b0 = *(gcv16bf*)(w0 + kb * 32);
    acc = wmma_bf16(a, b0, acc);
  }
  gf32* g = (gf32*)G;
  int rofs = (lane & 16) ? 8 : 0;
  int c0 = n0 + (lane & 15);
#pragma unroll
  for (int rr = 0; rr < 8; ++rr) g[(size_t)(mBase + rofs + rr) * ldG + c0] = acc[rr];
}

// ---------------- grid-wide barrier (persistent kernel) ----------------
__device__ inline void grid_barrier(unsigned* cnt, unsigned expected) {
  __syncthreads();
  if (threadIdx.x == 0) {
    __threadfence();
    __hip_atomic_fetch_add(cnt, 1u, __ATOMIC_ACQ_REL, __HIP_MEMORY_SCOPE_AGENT);
    while (__hip_atomic_load(cnt, __ATOMIC_ACQUIRE, __HIP_MEMORY_SCOPE_AGENT) < expected) {
      __builtin_amdgcn_s_sleep(2);
    }
    __threadfence();
  }
  __syncthreads();
}

// ---------------- phase bodies ----------------
// LSTM-1 elementwise update + attention window for one batch row (one wave).
__device__ inline void lstm1_window(int t, int b, int lane, float* swin,
                                    const float* gates1, const float* b0v, float* c1,
                                    const float* Ww, const float* bwv, float* kappa,
                                    const float* c_mask, const int* idxb, const float* inp,
                                    bf16_t* Acat1, bf16_t* Acat2, bf16_t* featB) {
  gcf32* gr  = (gcf32*)(gates1 + (size_t)b * G4H);
  gcf32* bg  = (gcf32*)b0v;
  gf32*  cg  = (gf32*)(c1 + (size_t)b * HID);
  gbf*   a1  = (gbf*)(Acat1 + (size_t)b * K1PAD);
  gbf*   a2  = (gbf*)(Acat2 + (size_t)b * K2PAD);
  gbf*   fb  = (gbf*)(featB + (size_t)b * KOUT);
  float hreg[13];
#pragma unroll
  for (int jj = 0; jj < 13; ++jj) {
    int j = lane + 32 * jj;
    float h = 0.f;
    if (j < HID) {
      float gi = gr[j]           + bg[j];
      float gf = gr[HID + j]     + bg[HID + j];
      float gg = gr[2 * HID + j] + bg[2 * HID + j];
      float go = gr[3 * HID + j] + bg[3 * HID + j];
      float cold = cg[j];
      float cn = sigmf(gf) * cold + sigmf(gi) * tanhf(gg);
      cg[j] = cn;
      h = sigmf(go) * tanhf(cn);
      bf16_t hb = f2bf(h);
      a1[60 + j] = hb;      // for gates1(t+1)
      a2[j]      = hb;      // for gates2(t)
      fb[j]      = hb;      // for output head(t)
    }
    hreg[jj] = h;
  }
  // wp = exp(h1 @ Ww^T + bw): 30 cross-lane dot reductions of length 400
  gcf32* Wwg = (gcf32*)Ww;
  gcf32* bwg = (gcf32*)bwv;
  float wpv[30];
#pragma unroll
  for (int m = 0; m < 30; ++m) {
    float p = 0.f;
#pragma unroll
    for (int jj = 0; jj < 13; ++jj) {
      int j = lane + 32 * jj;
      if (j < HID) p += hreg[jj] * Wwg[m * HID + j];
    }
#pragma unroll
    for (int o = 16; o > 0; o >>= 1) p += __shfl_xor(p, o, 32);  // all lanes get the sum
    wpv[m] = __expf(p + bwg[m]);
  }
  gf32* kpg = (gf32*)(kappa + (size_t)b * KMIX);
  float kap[KMIX];
#pragma unroll
  for (int k = 0; k < KMIX; ++k) {
    float kn = kpg[k] + wpv[20 + k] * KFC;
    kap[k] = kn;
    if (lane == 0) kpg[k] = kn;
  }
  // win[c] = sum_u phi[u]*mask[u]*[idx==c] via LDS scatter (c_seq is one-hot)
  for (int c = lane; c < 64; c += 32) swin[c] = 0.f;
  asm volatile("s_wait_dscnt 0x0" ::: "memory");
  gcf32* cmg = (gcf32*)(c_mask + (size_t)b * ULEN);
  gci32* idg = (gci32*)(idxb + (size_t)b * ULEN);
#pragma unroll
  for (int uu = 0; uu < 3; ++uu) {
    int u = lane + 32 * uu;
    if (u < ULEN) {
      float ph = 0.f;
#pragma unroll
      for (int k = 0; k < KMIX; ++k) {
        float d = kap[k] - (float)(u + 1);
        ph += wpv[k] * __expf(-wpv[10 + k] * d * d);
      }
      ph *= cmg[u];
      atomicAdd(&swin[idg[u]], ph);
    }
  }
  asm volatile("s_wait_dscnt 0x0" ::: "memory");
  for (int c = lane; c < NCHARS; c += 32) {
    bf16_t wb = f2bf(swin[c]);
    a1[3 + c]   = wb;     // win for step t+1 (layer 1)
    a2[403 + c] = wb;     // win for step t   (layer 2)
  }
  gcf32* ing = (gcf32*)inp;
  if (lane < 3) {
    a2[400 + lane] = f2bf(ing[((size_t)t * BATCH + b) * 3 + lane]);          // x_t
    if (t + 1 < T_STEPS)
      a1[lane] = f2bf(ing[((size_t)(t + 1) * BATCH + b) * 3 + lane]);        // x_{t+1}
  }
}

__device__ inline void lstm2_update(int b, int lane, const float* gates2, const float* b1v,
                                    float* c2, bf16_t* Acat2, bf16_t* featB) {
  gcf32* gr = (gcf32*)(gates2 + (size_t)b * G4H);
  gcf32* bg = (gcf32*)b1v;
  gf32*  cg = (gf32*)(c2 + (size_t)b * HID);
  gbf*   a2 = (gbf*)(Acat2 + (size_t)b * K2PAD);
  gbf*   fb = (gbf*)(featB + (size_t)b * KOUT);
#pragma unroll
  for (int jj = 0; jj < 13; ++jj) {
    int j = lane + 32 * jj;
    if (j < HID) {
      float gi = gr[j]           + bg[j];
      float gf = gr[HID + j]     + bg[HID + j];
      float gg = gr[2 * HID + j] + bg[2 * HID + j];
      float go = gr[3 * HID + j] + bg[3 * HID + j];
      float cold = cg[j];
      float cn = sigmf(gf) * cold + sigmf(gi) * tanhf(gg);
      cg[j] = cn;
      float h = sigmf(go) * tanhf(cn);
      bf16_t hb = f2bf(h);
      a2[460 + j] = hb;
      fb[400 + j] = hb;
    }
  }
}

// Output head for timestep tt, executed by one WG (r = 0..3 handles batch rows 32r..32r+31).
__device__ inline void output_head(int tt, int r, int wv, int lane, int tid,
                                   const bf16_t* featB, const bf16_t* Woutc, const float* boutv,
                                   float* out, float (*s_y)[132]) {
  int mt  = 2 * r + (wv >> 2);                 // M tile (16 batch rows)
  int nt0 = (wv & 3) * 2, nt1 = nt0 + 1;       // two N tiles, shared A fragments
  gcbf* ap = (gcbf*)featB + (size_t)(mt * 16 + (lane & 15)) * KOUT + ((lane & 16) ? 16 : 0);
  gcbf* w0 = (gcbf*)Woutc + (size_t)(nt0 * 16 + (lane & 15)) * KOUT + ((lane & 16) ? 16 : 0);
  gcbf* w1 = (gcbf*)Woutc + (size_t)(nt1 * 16 + (lane & 15)) * KOUT + ((lane & 16) ? 16 : 0);
  v8f acc0 = {}, acc1 = {};
  for (int kb = 0; kb < KOUT / 32; ++kb) {
    v16bf a  = load_a(ap, kb);
    v16bf b0 = *(gcv16bf*)(w0 + kb * 32);
    v16bf b1 = *(gcv16bf*)(w1 + kb * 32);
    acc0 = wmma_bf16(a, b0, acc0);
    acc1 = wmma_bf16(a, b1, acc1);
  }
  int lm = (wv >> 2) * 16 + ((lane & 16) ? 8 : 0);
#pragma unroll
  for (int rr = 0; rr < 8; ++rr) {
    s_y[lm + rr][nt0 * 16 + (lane & 15)] = acc0[rr];
    s_y[lm + rr][nt1 * 16 + (lane & 15)] = acc1[rr];
  }
  __syncthreads();
  if (tid < 32) {
    int b = 32 * r + tid;
    const float* yr = s_y[tid];
    gcf32* bo = (gcf32*)boutv;
    gf32* og = (gf32*)out + ((size_t)tt * BATCH + b) * YCOLS;
    og[0] = 1.f / (1.f + __expf(-(yr[120] + bo[120])));                    // e
    float mx = -3.4e38f;
    for (int c = 0; c < NGAUSS; ++c) mx = fmaxf(mx, yr[c] + bo[c]);
    float s = 0.f;
    for (int c = 0; c < NGAUSS; ++c) s += __expf(yr[c] + bo[c] - mx);
    float lse = mx + __logf(s);
    for (int c = 0; c < NGAUSS; ++c)     og[1 + c]              = yr[c] + bo[c] - lse;                       // log_pi
    for (int c = 0; c < 2 * NGAUSS; ++c) og[1 + NGAUSS + c]     = yr[NGAUSS + c] + bo[NGAUSS + c];           // mu
    for (int c = 0; c < 2 * NGAUSS; ++c) og[1 + 3 * NGAUSS + c] = __expf(yr[3 * NGAUSS + c] + bo[3 * NGAUSS + c]); // sigma
    for (int c = 0; c < NGAUSS; ++c)     og[1 + 5 * NGAUSS + c] = tanhf(yr[5 * NGAUSS + c] + bo[5 * NGAUSS + c]);  // rho
  }
}

// ---------------- prep kernel: pack bf16 weights, init state, one-hot -> indices ----------------
__global__ void hwrnn_prep(const float* inp, const float* c_seq,
                           const float* W_ih0, const float* W_hh0,
                           const float* W_ih1, const float* W_hh1, const float* Wout,
                           bf16_t* W1c, bf16_t* W2c, bf16_t* Woutc,
                           bf16_t* Acat1, bf16_t* Acat2, bf16_t* featB,
                           float* c1, float* c2, float* kappa, int* idxb) {
  const int S0 = G4H * K1PAD;
  const int S1 = S0 + G4H * K2PAD;
  const int S2 = S1 + 128 * KOUT;
  const int S3 = S2 + BATCH * K1PAD;
  const int S4 = S3 + BATCH * K2PAD;
  const int S5 = S4 + BATCH * KOUT;
  const int S6 = S5 + BATCH * HID;
  const int S7 = S6 + BATCH * HID;
  const int S8 = S7 + BATCH * KMIX;
  const int S9 = S8 + BATCH * ULEN;
  for (int i = blockIdx.x * blockDim.x + threadIdx.x; i < S9; i += gridDim.x * blockDim.x) {
    if (i < S0) {                                   // W1c: [W_ih0(x,win) | W_hh0 | 0]
      int n = i / K1PAD, k = i % K1PAD;
      float v = 0.f;
      if (k < 60) v = W_ih0[n * 60 + k];
      else if (k < 460) v = W_hh0[n * HID + (k - 60)];
      W1c[i] = f2bf(v);
    } else if (i < S1) {                            // W2c: [W_ih1(h1,x,win) | W_hh1 | 0]
      int j = i - S0, n = j / K2PAD, k = j % K2PAD;
      float v = 0.f;
      if (k < 460) v = W_ih1[n * 460 + k];
      else if (k < 860) v = W_hh1[n * HID + (k - 460)];
      W2c[j] = f2bf(v);
    } else if (i < S2) {                            // Wout padded to 128 rows
      int j = i - S1, n = j / KOUT, k = j % KOUT;
      Woutc[j] = f2bf(n < YCOLS ? Wout[n * KOUT + k] : 0.f);
    } else if (i < S3) {                            // Acat1: zeros except x_0
      int j = i - S2, b = j / K1PAD, k = j % K1PAD;
      Acat1[j] = (k < 3) ? f2bf(inp[b * 3 + k]) : f2bf(0.f);
    } else if (i < S4) { Acat2[i - S3] = f2bf(0.f); }
    else if (i < S5)   { featB[i - S4] = f2bf(0.f); }
    else if (i < S6)   { c1[i - S5] = 0.f; }
    else if (i < S7)   { c2[i - S6] = 0.f; }
    else if (i < S8)   { kappa[i - S7] = 0.f; }
    else {                                          // argmax of one-hot rows
      int j = i - S8;
      const float* p = c_seq + (size_t)j * NCHARS;
      int best = 0; float bv = p[0];
      for (int c = 1; c < NCHARS; ++c) { float v = p[c]; if (v > bv) { bv = v; best = c; } }
      idxb[j] = best;
    }
  }
}

// ---------------- persistent scan kernel ----------------
__global__ __launch_bounds__(256) void hwrnn_main(
    const float* __restrict__ inp, const float* __restrict__ c_mask,
    const float* __restrict__ b0v, const float* __restrict__ b1v,
    const float* __restrict__ Ww, const float* __restrict__ bwv, const float* __restrict__ boutv,
    const bf16_t* __restrict__ W1c, const bf16_t* __restrict__ W2c, const bf16_t* __restrict__ Woutc,
    bf16_t* __restrict__ Acat1, bf16_t* __restrict__ Acat2, bf16_t* __restrict__ featB,
    float* __restrict__ gates1, float* __restrict__ gates2,
    float* __restrict__ c1, float* __restrict__ c2, float* __restrict__ kappa,
    const int* __restrict__ idxb, unsigned* __restrict__ barCnt, float* __restrict__ out) {
  __shared__ float s_win[8][64];
  __shared__ float s_y[32][132];
  const int wg = blockIdx.x, tid = threadIdx.x;
  const int lane = tid & 31, wv = tid >> 5;
  const int gw = wg * 8 + wv;              // global wave id (0..255)
  const int TT = 8 * (G4H / 16);           // 800 C tiles per gate GEMM
  unsigned epoch = 0;

  for (int t = 0; t < T_STEPS; ++t) {
    // ---- slot A: gates1(t) on WGs [0,GATE_WGS); output head(t-1) on the rest ----
    if (wg < GATE_WGS) {
      const int STR = GATE_WGS * 8;        // 224 waves; stride preserves tile&7 (M tile)
      for (int tile = gw; tile < TT; tile += 2 * STR) {
        int tb = tile + STR;
        int mB = (tile & 7) * 16;
        if (tb < TT)
          gemm_pair(Acat1, K1PAD, W1c, K1PAD, gates1, G4H, mB,
                    (tile >> 3) * 16, (tb >> 3) * 16, K1PAD / 32, lane);
        else
          gemm_one(Acat1, K1PAD, W1c, K1PAD, gates1, G4H, mB,
                   (tile >> 3) * 16, K1PAD / 32, lane);
      }
    } else if (t > 0) {
      output_head(t - 1, wg - GATE_WGS, wv, lane, tid, featB, Woutc, boutv, out, s_y);
    }
    grid_barrier(barCnt, ++epoch * NWG);

    // ---- P1b: LSTM-1 elementwise + attention window, one wave per batch row ----
    if (gw < BATCH)
      lstm1_window(t, gw, lane, &s_win[wv][0], gates1, b0v, c1, Ww, bwv, kappa,
                   c_mask, idxb, inp, Acat1, Acat2, featB);
    grid_barrier(barCnt, ++epoch * NWG);

    // ---- slot C: gates2(t) on all WGs ----
    {
      const int STR = NWG * 8;             // 256 waves
      for (int tile = gw; tile < TT; tile += 2 * STR) {
        int tb = tile + STR;
        int mB = (tile & 7) * 16;
        if (tb < TT)
          gemm_pair(Acat2, K2PAD, W2c, K2PAD, gates2, G4H, mB,
                    (tile >> 3) * 16, (tb >> 3) * 16, K2PAD / 32, lane);
        else
          gemm_one(Acat2, K2PAD, W2c, K2PAD, gates2, G4H, mB,
                   (tile >> 3) * 16, K2PAD / 32, lane);
      }
    }
    grid_barrier(barCnt, ++epoch * NWG);

    // ---- P3b: LSTM-2 elementwise ----
    if (gw < BATCH) lstm2_update(gw, lane, gates2, b1v, c2, Acat2, featB);
    grid_barrier(barCnt, ++epoch * NWG);
  }
  // epilogue: output head for the final timestep
  if (wg >= GATE_WGS)
    output_head(T_STEPS - 1, wg - GATE_WGS, wv, lane, tid, featB, Woutc, boutv, out, s_y);
}

// ---------------- launcher ----------------
extern "C" void kernel_launch(void* const* d_in, const int* in_sizes, int n_in,
                              void* d_out, int out_size, void* d_ws, size_t ws_size,
                              hipStream_t stream) {
  const float* inp    = (const float*)d_in[0];
  const float* c_seq  = (const float*)d_in[1];
  const float* c_mask = (const float*)d_in[2];
  const float* W_ih0  = (const float*)d_in[3];
  const float* W_hh0  = (const float*)d_in[4];
  const float* b0v    = (const float*)d_in[5];
  const float* W_ih1  = (const float*)d_in[6];
  const float* W_hh1  = (const float*)d_in[7];
  const float* b1v    = (const float*)d_in[8];
  const float* Ww     = (const float*)d_in[9];
  const float* bwv    = (const float*)d_in[10];
  const float* Wout   = (const float*)d_in[11];
  const float* boutv  = (const float*)d_in[12];
  float* out = (float*)d_out;

  char* ws = (char*)d_ws;
  size_t o = 0;
  auto carve = [&](size_t bytes) -> char* {
    char* p = ws + o;
    o += (bytes + 255) & ~((size_t)255);
    return p;
  };
  bf16_t* W1c    = (bf16_t*)carve((size_t)G4H * K1PAD * 2);
  bf16_t* W2c    = (bf16_t*)carve((size_t)G4H * K2PAD * 2);
  bf16_t* Woutc  = (bf16_t*)carve((size_t)128 * KOUT * 2);
  bf16_t* Acat1  = (bf16_t*)carve((size_t)BATCH * K1PAD * 2);
  bf16_t* Acat2  = (bf16_t*)carve((size_t)BATCH * K2PAD * 2);
  bf16_t* featB  = (bf16_t*)carve((size_t)BATCH * KOUT * 2);
  float*  gates1 = (float*) carve((size_t)BATCH * G4H * 4);
  float*  gates2 = (float*) carve((size_t)BATCH * G4H * 4);
  float*  c1     = (float*) carve((size_t)BATCH * HID * 4);
  float*  c2     = (float*) carve((size_t)BATCH * HID * 4);
  float*  kappa  = (float*) carve((size_t)BATCH * KMIX * 4);
  int*    idxb   = (int*)   carve((size_t)BATCH * ULEN * 4);
  unsigned* barCnt = (unsigned*)carve(256);

  hipMemsetAsync(barCnt, 0, 256, stream);
  hwrnn_prep<<<1024, 256, 0, stream>>>(inp, c_seq, W_ih0, W_hh0, W_ih1, W_hh1, Wout,
                                       W1c, W2c, Woutc, Acat1, Acat2, featB,
                                       c1, c2, kappa, idxb);
  hwrnn_main<<<NWG, 256, 0, stream>>>(inp, c_mask, b0v, b1v, Ww, bwv, boutv,
                                      W1c, W2c, Woutc, Acat1, Acat2, featB,
                                      gates1, gates2, c1, c2, kappa, idxb, barCnt, out);
}
